// Decoder_47210280517684
// MI455X (gfx1250) — matrix operand order repaired and checked by reading, not compiled, not run on hardware
//
#include <hip/hip_runtime.h>
#include <math.h>

#define B_   128
#define T_   50
#define TM   49
#define V_   10000
#define VP   10240   // V_ padded to 640 N-tiles of 16
#define D_   512
#define E_   2048
#define G4D  2048    // 4*D

typedef __attribute__((ext_vector_type(16))) __bf16 v16bf;
typedef __attribute__((ext_vector_type(8)))  float  v8f;
typedef __attribute__((ext_vector_type(4)))  unsigned int u32x4;
typedef __attribute__((ext_vector_type(4)))  int i32x4;
typedef __attribute__((ext_vector_type(8)))  int i32x8;

#if defined(__AMDGCN__) && __has_builtin(__builtin_amdgcn_tensor_load_to_lds) && \
    __has_builtin(__builtin_amdgcn_s_wait_tensorcnt)
#define USE_TDM 1
#endif

union BFV { v16bf v; unsigned short u[16]; uint4 q[2]; };

__device__ __forceinline__ unsigned short f2bf(float x) {
    unsigned int b = __float_as_uint(x);
    b += 0x7FFFu + ((b >> 16) & 1u);   // round-to-nearest-even
    return (unsigned short)(b >> 16);
}

__device__ __forceinline__ v8f bfmma(v16bf a, v16bf b, v8f c) {
    return __builtin_amdgcn_wmma_f32_16x16x32_bf16(false, a, false, b,
                                                   (short)0, c, false, false);
}

// B fragment: weights row-major (N x K) in bf16, tile (n0,k0).
// lane = {n = lane&15, half = lane>>4}; groups of 8 K at k0+8*half and k0+16+8*half.
__device__ __forceinline__ v16bf load_b_frag(const unsigned short* W, int ldk,
                                             int n0, int k0, int lane) {
    int n = lane & 15, half = lane >> 4;
    const unsigned short* row = W + (size_t)(n0 + n) * ldk;
    const uint4* p0 = (const uint4*)(row + k0 + 8 * half);
    const uint4* p1 = (const uint4*)(row + k0 + 16 + 8 * half);
    BFV b; b.q[0] = *p0; b.q[1] = *p1;
    return b.v;
}

// A fragment from bf16 row-major (M x K), tile (m0,k0).
// lane row m = lane&15; groups of 8 at k0+8*half and k0+16+8*half.
__device__ __forceinline__ v16bf load_a_frag_bf(const unsigned short* A, int ldk,
                                                int m0, int k0, int lane) {
    int m = lane & 15, half = lane >> 4;
    const unsigned short* row = A + (size_t)(m0 + m) * ldk;
    const uint4* p0 = (const uint4*)(row + k0 + 8 * half);
    const uint4* p1 = (const uint4*)(row + k0 + 16 + 8 * half);
    BFV a; a.q[0] = *p0; a.q[1] = *p1;
    return a.v;
}

// A fragment converted on the fly from an f32 row (embedding gather path).
__device__ __forceinline__ v16bf load_a_frag_f32(const float* Arow, int k0, int lane) {
    int half = lane >> 4;
    const float4* g0 = (const float4*)(Arow + k0 + 8 * half);
    const float4* g1 = (const float4*)(Arow + k0 + 16 + 8 * half);
    float4 f0 = g0[0], f1 = g0[1], f2 = g1[0], f3 = g1[1];
    BFV a;
    a.u[0]=f2bf(f0.x); a.u[1]=f2bf(f0.y); a.u[2]=f2bf(f0.z); a.u[3]=f2bf(f0.w);
    a.u[4]=f2bf(f1.x); a.u[5]=f2bf(f1.y); a.u[6]=f2bf(f1.z); a.u[7]=f2bf(f1.w);
    a.u[8]=f2bf(f2.x); a.u[9]=f2bf(f2.y); a.u[10]=f2bf(f2.z); a.u[11]=f2bf(f2.w);
    a.u[12]=f2bf(f3.x); a.u[13]=f2bf(f3.y); a.u[14]=f2bf(f3.z); a.u[15]=f2bf(f3.w);
    return a.v;
}

// ---------------- setup kernels ----------------

__global__ void prep_kernel(const int* caplens, const int* captions,
                            const float* b_ih, const float* b_hh,
                            float* out_caps, float* out_dlen, float* out_sidx,
                            int* ws_sidx, int* ws_dlen, int* ws_caps, float* ws_bias) {
    __shared__ int cl_s[B_];
    __shared__ int sidx_s[B_];
    int b = threadIdx.x;
    if (b < B_) cl_s[b] = caplens[b];
    __syncthreads();
    if (b < B_) {
        int myc = cl_s[b], rank = 0;
        for (int j = 0; j < B_; ++j) {
            int cj = cl_s[j];
            if (cj > myc || (cj == myc && j < b)) rank++;   // stable descending
        }
        sidx_s[rank] = b;
    }
    __syncthreads();
    if (b < B_) {
        int src = sidx_s[b];
        ws_sidx[b]  = src;
        out_sidx[b] = (float)src;
        int dl = cl_s[src] - 1;
        ws_dlen[b]  = dl;
        out_dlen[b] = (float)dl;
        for (int t = 0; t < T_; ++t) {
            int tok = captions[src * T_ + t];
            ws_caps[b * T_ + t]  = tok;
            out_caps[b * T_ + t] = (float)tok;
        }
    }
    for (int n = threadIdx.x; n < G4D; n += blockDim.x)
        ws_bias[n] = b_ih[n] + b_hh[n];
}

__global__ void cvt_bf16_kernel(const float* src, unsigned short* dst, long n) {
    long i = (long)blockIdx.x * blockDim.x + threadIdx.x;
    long stride = (long)gridDim.x * blockDim.x;
    for (; i < n; i += stride) dst[i] = f2bf(src[i]);
}

__global__ void zfill_kernel(unsigned short* dst, long n) {
    long i = (long)blockIdx.x * blockDim.x + threadIdx.x;
    long stride = (long)gridDim.x * blockDim.x;
    for (; i < n; i += stride) dst[i] = 0;
}

__global__ void gather_enc_kernel(const float* enc, const int* sidx,
                                  unsigned short* enc_bf) {
    int r = blockIdx.x;
    int src = sidx[r];
    for (int i = threadIdx.x; i < E_; i += blockDim.x)
        enc_bf[(size_t)r * E_ + i] = f2bf(enc[(size_t)src * E_ + i]);
}

// ---------------- GEMM kernels (16x16 WMMA tiles) ----------------

// h0 = enc_s @ init_h_W.T + init_h_b ; c0 likewise (z selects).
__global__ void init_state_kernel(const unsigned short* enc_bf,
                                  const unsigned short* Wh_bf,
                                  const unsigned short* Wc_bf,
                                  const float* hb, const float* cb,
                                  float* h, float* c, unsigned short* h_bf) {
    int lane = threadIdx.x & 31;
    int m0 = blockIdx.x * 16, n0 = blockIdx.y * 16;
    bool is_c = (blockIdx.z != 0);
    const unsigned short* W = is_c ? Wc_bf : Wh_bf;
    const float* bias = is_c ? cb : hb;
    v8f acc = {};
    for (int k0 = 0; k0 < E_; k0 += 32) {
        v16bf a  = load_a_frag_bf(enc_bf, E_, m0, k0, lane);
        v16bf bb = load_b_frag(W, E_, n0, k0, lane);
        acc = bfmma(a, bb, acc);
    }
    int n = lane & 15, half = lane >> 4;
    float bn = bias[n0 + n];
#pragma unroll
    for (int r = 0; r < 8; ++r) {
        int row = m0 + 8 * half + r;
        int col = n0 + n;
        float v = acc[r] + bn;
        if (is_c) c[(size_t)row * D_ + col] = v;
        else {
            h[(size_t)row * D_ + col]    = v;
            h_bf[(size_t)row * D_ + col] = f2bf(v);
        }
    }
}

// gates_x[t*B+b, :] = emb_W[tok(b,t)] @ W_ih.T + (b_ih + b_hh) for all 49 steps.
__global__ void xgates_kernel(const float* emb_W, const int* ws_caps,
                              const unsigned short* Wih_bf, const float* bias,
                              float* gates_x) {
    int wave = threadIdx.x >> 5, lane = threadIdx.x & 31;
    int p0 = blockIdx.x * 16;                 // batched-row tile (p = t*128+b)
    int n0 = (blockIdx.y * 8 + wave) * 16;
    int m = lane & 15, half = lane >> 4, n = lane & 15;
    int p = p0 + m;
    int t = p / B_, b = p % B_;
    const float* Arow = emb_W + (size_t)ws_caps[b * T_ + t] * D_;
    v8f acc = {};
    for (int k0 = 0; k0 < D_; k0 += 32) {
        v16bf a  = load_a_frag_f32(Arow, k0, lane);
        v16bf bb = load_b_frag(Wih_bf, D_, n0, k0, lane);
        acc = bfmma(a, bb, acc);
    }
    float bn = bias[n0 + n];
#pragma unroll
    for (int r = 0; r < 8; ++r) {
        int prow = p0 + 8 * half + r;
        gates_x[(size_t)prow * G4D + n0 + n] = acc[r] + bn;
    }
}

// gates = h_bf @ W_hh.T + gates_x[t]
__global__ void hgates_kernel(const unsigned short* h_bf,
                              const unsigned short* Whh_bf,
                              const float* gates_x, float* gates, int t) {
    int wave = threadIdx.x >> 5, lane = threadIdx.x & 31;
    int m0 = blockIdx.x * 16;
    int n0 = (blockIdx.y * 8 + wave) * 16;
    v8f acc = {};
    for (int k0 = 0; k0 < D_; k0 += 32) {
        v16bf a  = load_a_frag_bf(h_bf, D_, m0, k0, lane);
        v16bf bb = load_b_frag(Whh_bf, D_, n0, k0, lane);
        acc = bfmma(a, bb, acc);
    }
    int n = lane & 15, half = lane >> 4;
#pragma unroll
    for (int r = 0; r < 8; ++r) {
        int row = m0 + 8 * half + r;
        size_t col = (size_t)(n0 + n);
        gates[(size_t)row * G4D + col] =
            acc[r] + gates_x[((size_t)t * B_ + row) * G4D + col];
    }
}

__global__ void cell_kernel(const float* gates, const int* dlen,
                            float* h, float* c, unsigned short* h_bf, int t) {
    int idx = blockIdx.x * blockDim.x + threadIdx.x;   // idx = b*512 + d
    int b = idx / D_, d = idx % D_;
    const float* g = gates + (size_t)b * G4D;
    float ig = g[d], fg = g[D_ + d], gg = g[2 * D_ + d], og = g[3 * D_ + d];
    float i = 1.f / (1.f + __expf(-ig));
    float f = 1.f / (1.f + __expf(-fg));
    float gt = tanhf(gg);
    float o = 1.f / (1.f + __expf(-og));
    float cn = f * c[idx] + i * gt;
    float hn = o * tanhf(cn);
    if (t < dlen[b]) {
        c[idx] = cn;
        h[idx] = hn;
        h_bf[idx] = f2bf(hn);
    }
}

// preds[:,t,:] = mask ? h @ fc_W.T + fc_b : 0
// fcW_bf is padded to 640 N-tiles so the inner loop is branch-free:
// per k-step 1 A-load (LDS, TDM-staged) + 4 B-loads + 4 WMMAs.
__global__ void fc_kernel(const unsigned short* h_bf, const unsigned short* fcW_bf,
                          const float* fc_b, const int* dlen, float* preds, int t) {
    __shared__ unsigned short Asm[16 * D_];
    int m0 = blockIdx.x * 16;
#ifdef USE_TDM
    if (threadIdx.x < 32) {
        // 2D D#: data_size=2B, tile 512x16, tensor_dim0=512, dim0_stride=512.
        unsigned long long ga = (unsigned long long)(const void*)(h_bf + (size_t)m0 * D_);
        unsigned lds = (unsigned)(unsigned long long)(const void*)&Asm[0];
        u32x4 g0 = { 1u,                               // count=1, is_restore=0
                     lds,                              // lds_addr
                     (unsigned)ga,                     // global_addr[31:0]
                     ((unsigned)(ga >> 32) & 0x01FFFFFFu) | (2u << 30) }; // [56:32] | type=2
        i32x8 g1 = { (int)0x10000,                     // data_size=1 (2 bytes)
                     (int)(512u << 16),                // tensor_dim0 = 512
                     (int)(16u << 16),                 // tensor_dim1 = 16
                     (int)(512u << 16),                // tile_dim0 = 512
                     16,                               // tile_dim1 = 16
                     512,                              // tensor_dim0_stride = 512
                     0, 0 };
        i32x4 gz = { 0, 0, 0, 0 };
#if __clang_major__ >= 23
        i32x8 gz8 = { 0, 0, 0, 0, 0, 0, 0, 0 };
        __builtin_amdgcn_tensor_load_to_lds(g0, g1, gz, gz, gz8, 0);
#else
        __builtin_amdgcn_tensor_load_to_lds(g0, g1, gz, gz, 0);
#endif
        __builtin_amdgcn_s_wait_tensorcnt(0);
    }
#else
    {
        const uint4* gsrc = (const uint4*)(h_bf + (size_t)m0 * D_);
        uint4* ldst = (uint4*)Asm;
        for (int i = threadIdx.x; i < (16 * D_) / 8; i += blockDim.x) ldst[i] = gsrc[i];
    }
#endif
    __syncthreads();

    int wave = threadIdx.x >> 5, lane = threadIdx.x & 31;
    int ntb = (blockIdx.y * 8 + wave) * 4;        // 0..636, always valid in padded W
    v8f acc[4] = {v8f{}, v8f{}, v8f{}, v8f{}};
    for (int k0 = 0; k0 < D_; k0 += 32) {
        v16bf a = load_a_frag_bf(Asm, D_, 0, k0, lane);   // LDS reads
#pragma unroll
        for (int j = 0; j < 4; ++j) {
            v16bf bb = load_b_frag(fcW_bf, D_, (ntb + j) * 16, k0, lane);
            acc[j] = bfmma(a, bb, acc[j]);
        }
    }
    int n = lane & 15, half = lane >> 4;
#pragma unroll
    for (int j = 0; j < 4; ++j) {
        int nt = ntb + j;
        if (nt >= V_ / 16) continue;              // discard pad tiles (store only)
        int n0 = nt * 16;
        float bn = fc_b[n0 + n];
#pragma unroll
        for (int r = 0; r < 8; ++r) {
            int row = m0 + 8 * half + r;
            float v = (t < dlen[row]) ? (acc[j][r] + bn) : 0.f;
            preds[(size_t)row * TM * V_ + (size_t)t * V_ + (n0 + n)] = v;
        }
    }
}

// ---------------- host launch ----------------

extern "C" void kernel_launch(void* const* d_in, const int* in_sizes, int n_in,
                              void* d_out, int out_size, void* d_ws, size_t ws_size,
                              hipStream_t stream) {
    const float* enc_out  = (const float*)d_in[0];
    const int*   captions = (const int*)  d_in[1];
    const int*   caplens  = (const int*)  d_in[2];
    const float* emb_W    = (const float*)d_in[3];
    const float* W_ih     = (const float*)d_in[4];
    const float* W_hh     = (const float*)d_in[5];
    const float* b_ih     = (const float*)d_in[6];
    const float* b_hh     = (const float*)d_in[7];
    const float* fc_W     = (const float*)d_in[8];
    const float* fc_b     = (const float*)d_in[9];
    const float* init_h_W = (const float*)d_in[10];
    const float* init_h_b = (const float*)d_in[11];
    const float* init_c_W = (const float*)d_in[12];
    const float* init_c_b = (const float*)d_in[13];

    float* out = (float*)d_out;
    float* out_preds = out;                                   // 128*49*10000
    float* out_caps  = out + (size_t)B_ * TM * V_;            // 128*50
    float* out_dlen  = out_caps + B_ * T_;                    // 128
    float* out_sidx  = out_dlen + B_;                         // 128

    char* base = (char*)d_ws;
    size_t off = 0;
#define WALLOC(ptr, type, count)                                               \
    type* ptr = (type*)(base + off);                                           \
    off += (((size_t)(count) * sizeof(type)) + 255) & ~(size_t)255;

    WALLOC(ws_sidx, int, B_)
    WALLOC(ws_dlen, int, B_)
    WALLOC(ws_caps, int, B_ * T_)
    WALLOC(ws_bias, float, G4D)
    WALLOC(enc_bf,  unsigned short, B_ * E_)
    WALLOC(Wih_bf,  unsigned short, (size_t)G4D * D_)
    WALLOC(Whh_bf,  unsigned short, (size_t)G4D * D_)
    WALLOC(fcW_bf,  unsigned short, (size_t)VP * D_)          // padded to 640 tiles
    WALLOC(iHW_bf,  unsigned short, (size_t)D_ * E_)
    WALLOC(iCW_bf,  unsigned short, (size_t)D_ * E_)
    WALLOC(ws_h,    float, B_ * D_)
    WALLOC(ws_c,    float, B_ * D_)
    WALLOC(ws_h_bf, unsigned short, B_ * D_)
    WALLOC(ws_gates, float, (size_t)B_ * G4D)
    WALLOC(ws_gx,   float, (size_t)TM * B_ * G4D)
#undef WALLOC
    (void)ws_size; (void)n_in; (void)in_sizes; (void)out_size;

    prep_kernel<<<1, 128, 0, stream>>>(caplens, captions, b_ih, b_hh,
                                       out_caps, out_dlen, out_sidx,
                                       ws_sidx, ws_dlen, ws_caps, ws_bias);

    cvt_bf16_kernel<<<1024, 256, 0, stream>>>(W_ih, Wih_bf, (long)G4D * D_);
    cvt_bf16_kernel<<<1024, 256, 0, stream>>>(W_hh, Whh_bf, (long)G4D * D_);
    cvt_bf16_kernel<<<2048, 256, 0, stream>>>(fc_W, fcW_bf, (long)V_ * D_);
    zfill_kernel<<<256, 256, 0, stream>>>(fcW_bf + (size_t)V_ * D_,
                                          (long)(VP - V_) * D_);
    cvt_bf16_kernel<<<1024, 256, 0, stream>>>(init_h_W, iHW_bf, (long)D_ * E_);
    cvt_bf16_kernel<<<1024, 256, 0, stream>>>(init_c_W, iCW_bf, (long)D_ * E_);
    gather_enc_kernel<<<128, 256, 0, stream>>>(enc_out, ws_sidx, enc_bf);

    init_state_kernel<<<dim3(B_ / 16, D_ / 16, 2), 32, 0, stream>>>(
        enc_bf, iHW_bf, iCW_bf, init_h_b, init_c_b, ws_h, ws_c, ws_h_bf);

    // All 49 steps of input-side gates in one batched GEMM (no recurrence).
    xgates_kernel<<<dim3((TM * B_) / 16, G4D / (16 * 8)), 256, 0, stream>>>(
        emb_W, ws_caps, Wih_bf, ws_bias, ws_gx);

    for (int t = 0; t < TM; ++t) {
        hgates_kernel<<<dim3(B_ / 16, G4D / (16 * 8)), 256, 0, stream>>>(
            ws_h_bf, Whh_bf, ws_gx, ws_gates, t);
        cell_kernel<<<(B_ * D_) / 256, 256, 0, stream>>>(
            ws_gates, ws_dlen, ws_h, ws_c, ws_h_bf, t);
        fc_kernel<<<dim3(B_ / 16, VP / 16 / 32), 256, 0, stream>>>(
            ws_h_bf, fcW_bf, fc_b, ws_dlen, out_preds, t);
    }
}
